// FOVKPN_77859167142102
// MI455X (gfx1250) — matrix operand order repaired and checked by reading, not compiled, not run on hardware
//
#include <hip/hip_runtime.h>
#include <hip/hip_bf16.h>

// ---------------------------------------------------------------------------
// FOVKPN forward for MI455X (gfx1250, wave32).
// All convolutions (3x3 / 2x2-s2 / 1x1, and the deformable-conv contraction
// recast as a 1x1 conv over C*9 channels) run as implicit-GEMM on the WMMA
// pipe: bf16 A/B fragments, f32 accumulation, 32x32 macro-tile per wave
// (2x2 grid of 16x16 WMMA tiles -> 4 v_wmma per K-chunk).
// Weight tiles are staged LDS-side by the Tensor Data Mover (TDM): one
// tensor_load_to_lds per wave per K-chunk instead of 32 global loads,
// overlapped with the im2col gather of the B tile, synchronized with
// s_wait_tensorcnt.
// ---------------------------------------------------------------------------

typedef __bf16 v16bf __attribute__((ext_vector_type(16)));
typedef float  v8f   __attribute__((ext_vector_type(8)));
typedef unsigned int v4u __attribute__((ext_vector_type(4)));
typedef int          v8i __attribute__((ext_vector_type(8)));
typedef int          v4i __attribute__((ext_vector_type(4)));

static __device__ __forceinline__ int iclampi(int v, int lo, int hi) {
    return v < lo ? lo : (v > hi ? hi : v);
}

// ---------------------------------------------------------------------------
// TDM: DMA a 32(rows) x 32(cols) fp32 tile, row stride strideK elements, into
// LDS at byte address ldsAddr. remK/remOC are the remaining tensor extents
// from gptr; TDM zero-fills reads beyond them (K tail / oc tail handling).
// D# packing per CDNA5 ISA ch.8 (group0: flags/lds/global addr; group1:
// data_size, tensor dims, tile dims, dim0 stride; groups 2/3 unused for 2D).
// ---------------------------------------------------------------------------
static __device__ __forceinline__ void tdm_load_tile_f32(
    unsigned ldsAddr, const float* gptr, int remK, int remOC, int strideK)
{
    unsigned long long ga = (unsigned long long)(size_t)gptr;
    v4u g0 = { 1u,                                        // count=1, user D#
               ldsAddr,                                   // lds_addr
               (unsigned)(ga & 0xffffffffull),            // global_addr lo
               (unsigned)((ga >> 32) & 0x01ffffffull)     // global_addr hi
                   | (2u << 30) };                        // type = 2 (image)
    unsigned d0 = (unsigned)(remK  < 0 ? 0 : remK);       // tensor_dim0
    unsigned d1 = (unsigned)(remOC < 0 ? 0 : remOC);      // tensor_dim1
    unsigned st = (unsigned)strideK;                      // dim0 stride
    v8i g1 = { (int)(2u << 16),                           // data_size=4B
               (int)((d0 & 0xffffu) << 16),               // dim0[15:0]
               (int)((d0 >> 16) | ((d1 & 0xffffu) << 16)),// dim0[31:16],dim1[15:0]
               (int)((d1 >> 16) | (32u << 16)),           // dim1[31:16],tile_dim0=32
               (int)32,                                   // tile_dim1=32, tile_dim2=0
               (int)st,                                   // dim0_stride[31:0]
               0, 0 };                                    // stride hi / dim1_stride
    v4i gz = {0, 0, 0, 0};
#if defined(__clang_major__) && (__clang_major__ <= 22)
    __builtin_amdgcn_tensor_load_to_lds(g0, g1, gz, gz, 0);
#else
    v8i gz8 = {0, 0, 0, 0, 0, 0, 0, 0};
    __builtin_amdgcn_tensor_load_to_lds(g0, g1, gz, gz, gz8, 0);
#endif
}

// ---------------------------------------------------------------------------
// Generic implicit-GEMM conv, NCHW, batch 1, kernel size KS (compile-time).
//   out[oc, oy, ox] = act( bias[oc] + sum_{ci,kh,kw} in[ci,iy,ix]*w[oc,ci,kh,kw] )
//                     (+ resid[oc,oy,ox] if resid != nullptr)
// act: 0 = identity, 1 = leaky-relu(slope), 2 = sigmoid(leaky-relu(slope))
// One wave computes a 32(oc) x 32(px) macro-tile; K = Cin*KS*KS in chunks of 32.
// A tile arrives via TDM (fp32 in LDS, converted to bf16 at fragment build);
// B tile is an im2col gather with each lane owning a fixed pixel column.
// ---------------------------------------------------------------------------
template <int KS>
__global__ __launch_bounds__(128) void conv_wmma(
    const float* __restrict__ in, const float* __restrict__ w,
    const float* __restrict__ bias, const float* __restrict__ resid,
    float* __restrict__ out,
    int Cin, int Cout, int Hin, int Win, int Hout, int Wout,
    int stride, int pad, int act, float slope)
{
    constexpr int KSQ = KS * KS;
    const int lane = threadIdx.x & 31;
    const int wv   = threadIdx.x >> 5;                 // wave in block (0..3)

    __shared__ float  ldsAf[4][32 * 32];               // weights (fp32, TDM dest)
    __shared__ __bf16 ldsB [4][32 * 32];               // im2col: [k][px_local]

    const int npx     = Hout * Wout;
    const int pxTiles = (npx + 31) >> 5;
    const int tileIdx = blockIdx.x * 4 + wv;
    const bool active = tileIdx < pxTiles;
    const int px0     = (active ? tileIdx : 0) * 32;
    const int oc0     = blockIdx.y * 32;
    const int Ktot    = Cin * KSQ;

    const unsigned ldsAddrA = (unsigned)(size_t)&ldsAf[wv][0];

    // Per-lane fixed im2col column: pixel px0+lane (one divide per kernel).
    const int  pxL     = px0 + lane;
    const bool pxValid = pxL < npx;
    const int  oyL     = pxValid ? (pxL / Wout) : 0;
    const int  oxL     = pxValid ? (pxL - oyL * Wout) : 0;
    const int  baseY   = oyL * stride - pad;
    const int  baseX   = oxL * stride - pad;

    v8f acc00 = {0.f,0.f,0.f,0.f,0.f,0.f,0.f,0.f};
    v8f acc01 = acc00, acc10 = acc00, acc11 = acc00;

    for (int k0 = 0; k0 < Ktot; k0 += 32) {
        // ---- issue TDM for A tile: 32 oc x 32 k, fp32, zero-filled tails ---
        tdm_load_tile_f32(ldsAddrA, w + (size_t)oc0 * Ktot + k0,
                          Ktot - k0, Cout - oc0, Ktot);

        // ---- stage B tile while the DMA flies (lane owns pixel column) -----
        #pragma unroll 8
        for (int j = 0; j < 32; ++j) {
            int k = k0 + j;
            float v = 0.f;
            if (k < Ktot && pxValid) {
                int ci   = k / KSQ;                    // compile-time KSQ
                int rem  = k - ci * KSQ;
                int kh   = rem / KS;
                int kw2  = rem - kh * KS;
                int iy   = baseY + kh;
                int ix   = baseX + kw2;
                if (iy >= 0 && iy < Hin && ix >= 0 && ix < Win)
                    v = in[((size_t)ci * Hin + iy) * Win + ix];
            }
            ldsB[wv][j * 32 + lane] = (__bf16)v;
        }
        asm volatile("s_wait_dscnt 0" ::: "memory");
        asm volatile("s_wait_tensorcnt 0" ::: "memory");

        // ---- build fragments per ISA 7.12.2 (wave32) -----------------------
        // A 16x32: lanes 0-15 -> M=lane, K 0..7 & 16..23; lanes 16-31 -> K 8..15 & 24..31.
        // B 32x16: lane<16 -> col=lane, K 0..7 & 16..23; lane>=16 -> col=lane-16, K 8..15 & 24..31.
        const int halfL = lane >> 4;
        const int mL    = lane & 15;
        v16bf a0, a1, b0, b1;
        #pragma unroll
        for (int i = 0; i < 8; ++i) {
            a0[i]     = (__bf16)ldsAf[wv][mL * 32 + halfL * 8 + i];
            a0[i + 8] = (__bf16)ldsAf[wv][mL * 32 + 16 + halfL * 8 + i];
            a1[i]     = (__bf16)ldsAf[wv][(16 + mL) * 32 + halfL * 8 + i];
            a1[i + 8] = (__bf16)ldsAf[wv][(16 + mL) * 32 + 16 + halfL * 8 + i];
        }
        #pragma unroll
        for (int e = 0; e < 16; ++e) {
            b0[e] = ldsB[wv][(halfL * 8 + (e & 7) + (e >> 3) * 16) * 32 + mL];
            b1[e] = ldsB[wv][(halfL * 8 + (e & 7) + (e >> 3) * 16) * 32 + 16 + mL];
        }

        acc00 = __builtin_amdgcn_wmma_f32_16x16x32_bf16(false, a0, false, b0, (short)0, acc00, false, false);
        acc01 = __builtin_amdgcn_wmma_f32_16x16x32_bf16(false, a0, false, b1, (short)0, acc01, false, false);
        acc10 = __builtin_amdgcn_wmma_f32_16x16x32_bf16(false, a1, false, b0, (short)0, acc10, false, false);
        acc11 = __builtin_amdgcn_wmma_f32_16x16x32_bf16(false, a1, false, b1, (short)0, acc11, false, false);
    }

    if (!active) return;

    // ---- epilogue: bias + activation + optional residual, D layout ---------
    // D 16x16: lane<16 -> N=lane, M=r; lane>=16 -> N=lane-16, M=8+r.
    const int halfL = lane >> 4;
    const int nL    = lane & 15;
    #pragma unroll
    for (int h = 0; h < 2; ++h) {
        #pragma unroll
        for (int q = 0; q < 2; ++q) {
            const v8f acc = (h == 0) ? (q == 0 ? acc00 : acc01)
                                     : (q == 0 ? acc10 : acc11);
            int px = px0 + q * 16 + nL;
            if (px >= npx) continue;
            #pragma unroll
            for (int r = 0; r < 8; ++r) {
                int oc = oc0 + h * 16 + halfL * 8 + r;
                if (oc < Cout) {
                    float v = acc[r] + bias[oc];
                    if (act >= 1) v = (v >= 0.f) ? v : slope * v;
                    if (act == 2) v = 1.f / (1.f + __expf(-v));
                    if (resid) v += resid[(size_t)oc * npx + px];
                    out[(size_t)oc * npx + px] = v;
                }
            }
        }
    }
}

// ---------------------------------------------------------------------------
// conv_t2: 2x2 transposed conv (pixel-shuffle style upsample).
// out[co, 2y+i, 2x+j] = bias[co] + sum_c x[c,y,x] * w[c, co, i, j]
// ---------------------------------------------------------------------------
__global__ void conv_t2_k(const float* __restrict__ x, const float* __restrict__ w,
                          const float* __restrict__ b, float* __restrict__ out,
                          int Cin, int Cout, int H, int W)
{
    const int Ho = 2 * H, Wo = 2 * W;
    const int total = Cout * Ho * Wo;
    int idx = blockIdx.x * blockDim.x + threadIdx.x;
    if (idx >= total) return;
    int co  = idx / (Ho * Wo);
    int rem = idx - co * (Ho * Wo);
    int oy  = rem / Wo, ox = rem - oy * Wo;
    int y = oy >> 1, i = oy & 1;
    int xx = ox >> 1, j = ox & 1;
    float s = b[co];
    for (int c = 0; c < Cin; ++c)
        s += x[((size_t)c * H + y) * W + xx] *
             w[(((size_t)c * Cout + co) * 2 + i) * 2 + j];
    out[idx] = s;
}

// ---------------------------------------------------------------------------
// Bilinear 2x upsample (half-pixel centers, edge clamp), output scaled by gain.
// ---------------------------------------------------------------------------
__global__ void upsample2x_k(const float* __restrict__ x, float* __restrict__ out,
                             int C, int H, int W, float gain)
{
    const int Ho = 2 * H, Wo = 2 * W;
    const int total = C * Ho * Wo;
    int idx = blockIdx.x * blockDim.x + threadIdx.x;
    if (idx >= total) return;
    int c   = idx / (Ho * Wo);
    int rem = idx - c * (Ho * Wo);
    int oy  = rem / Wo, ox = rem - oy * Wo;
    float sy = (oy + 0.5f) * 0.5f - 0.5f;
    float sx = (ox + 0.5f) * 0.5f - 0.5f;
    sy = fminf(fmaxf(sy, 0.f), (float)(H - 1));
    sx = fminf(fmaxf(sx, 0.f), (float)(W - 1));
    int y0 = (int)sy, x0 = (int)sx;
    int y1 = y0 + 1 < H ? y0 + 1 : H - 1;
    int x1 = x0 + 1 < W ? x0 + 1 : W - 1;
    float ty = sy - y0, tx = sx - x0;
    const float* xp = x + (size_t)c * H * W;
    float v = (1.f - ty) * ((1.f - tx) * xp[y0 * W + x0] + tx * xp[y0 * W + x1]) +
              ty        * ((1.f - tx) * xp[y1 * W + x0] + tx * xp[y1 * W + x1]);
    out[idx] = gain * v;
}

// att = ai * sigmoid_branch + ai
__global__ void att_k(const float* __restrict__ ai, const float* __restrict__ af,
                      float* __restrict__ out, int n)
{
    int i = blockIdx.x * blockDim.x + threadIdx.x;
    if (i < n) { float a = ai[i]; out[i] = a * af[i] + a; }
}

// ---------------------------------------------------------------------------
// Deformable-conv sampling stage: produce val[c*9 + t, h, w] =
//   sigmoid(m[g,t]) * bilinear(x[c], (y+dy+oy, x+dx+ox)) with zero OOB.
// One thread per (group, tap, pixel); inner loop over cg channels in group.
// ---------------------------------------------------------------------------
__global__ void deform_sample_k(const float* __restrict__ x, const float* __restrict__ om,
                                float* __restrict__ val, int C, int H, int W, int G)
{
    const int HW = H * W;
    const int total = G * 9 * HW;
    int idx = blockIdx.x * blockDim.x + threadIdx.x;
    if (idx >= total) return;
    int p  = idx % HW;
    int gt = idx / HW;
    int t  = gt % 9, g = gt / 9;
    int py = p / W, px = p - py * W;
    const int b9 = G * 9;

    float oy = om[(size_t)(g * 9 + t) * HW + p];
    float ox = om[(size_t)(b9 + g * 9 + t) * HW + p];
    float mr = om[(size_t)(2 * b9 + g * 9 + t) * HW + p];
    float m  = 1.f / (1.f + __expf(-mr));

    float sy = (float)(py + t / 3 - 1) + oy;
    float sx = (float)(px + t % 3 - 1) + ox;
    float y0f = floorf(sy), x0f = floorf(sx);
    float ty = sy - y0f, tx = sx - x0f;
    int y0 = (int)y0f, x0 = (int)x0f;
    int y1 = y0 + 1,  x1 = x0 + 1;

    float w00 = (1.f - ty) * (1.f - tx), w01 = (1.f - ty) * tx;
    float w10 = ty * (1.f - tx),         w11 = ty * tx;
    if (!(y0 >= 0 && y0 < H && x0 >= 0 && x0 < W)) w00 = 0.f;
    if (!(y0 >= 0 && y0 < H && x1 >= 0 && x1 < W)) w01 = 0.f;
    if (!(y1 >= 0 && y1 < H && x0 >= 0 && x0 < W)) w10 = 0.f;
    if (!(y1 >= 0 && y1 < H && x1 >= 0 && x1 < W)) w11 = 0.f;
    int y0c = iclampi(y0, 0, H - 1), y1c = iclampi(y1, 0, H - 1);
    int x0c = iclampi(x0, 0, W - 1), x1c = iclampi(x1, 0, W - 1);
    int i00 = y0c * W + x0c, i01 = y0c * W + x1c;
    int i10 = y1c * W + x0c, i11 = y1c * W + x1c;

    const int cg = C / G;
    for (int ci = 0; ci < cg; ++ci) {
        int c = g * cg + ci;
        const float* xp = x + (size_t)c * HW;
        float v = w00 * xp[i00] + w01 * xp[i01] + w10 * xp[i10] + w11 * xp[i11];
        val[(size_t)(c * 9 + t) * HW + p] = m * v;
    }
}

// ---------------------------------------------------------------------------
// Kernel prediction: preds[chOff + c] = sum_{i,j} core[c*25+i*5+j] * img_pad
// (5x5 taps, dilation = rate, zero padding).
// ---------------------------------------------------------------------------
__global__ void kernel_pred_k(const float* __restrict__ img, const float* __restrict__ core,
                              float* __restrict__ preds, int H, int W, int rate, int chOff)
{
    const int HW = H * W;
    int idx = blockIdx.x * blockDim.x + threadIdx.x;
    if (idx >= 3 * HW) return;
    int c = idx / HW;
    int p = idx - c * HW;
    int y = p / W, x = p - y * W;
    float s = 0.f;
    #pragma unroll
    for (int i = 0; i < 5; ++i) {
        #pragma unroll
        for (int j = 0; j < 5; ++j) {
            int yy = y + (i - 2) * rate;
            int xx = x + (j - 2) * rate;
            float iv = (yy >= 0 && yy < H && xx >= 0 && xx < W)
                           ? img[((size_t)c * H + yy) * W + xx] : 0.f;
            s += core[((size_t)(c * 25 + i * 5 + j) * H + y) * W + x] * iv;
        }
    }
    preds[((size_t)(chOff + c) * H + y) * W + x] = s;
}

// ---------------------------------------------------------------------------
// Host orchestration
// ---------------------------------------------------------------------------
static inline int cdiv(int a, int b) { return (a + b - 1) / b; }

extern "C" void kernel_launch(void* const* d_in, const int* in_sizes, int n_in,
                              void* d_out, int out_size, void* d_ws, size_t ws_size,
                              hipStream_t stream)
{
    (void)in_sizes; (void)n_in; (void)out_size; (void)ws_size;
    const float* x = (const float*)d_in[0];
    auto Wp = [&](int i) { return (const float*)d_in[i]; };

    const int H1 = 384, W1 = 384, HW1 = H1 * W1;
    const int H2 = 192, W2 = 192, HW2 = H2 * W2;
    const int H3 = 96,  W3 = 96,  HW3 = H3 * W3;
    const int H4 = 48,  W4 = 48,  HW4 = H4 * W4;

    size_t off = 0;
    auto alloc = [&](size_t nflt) -> float* {
        float* p = (float*)((char*)d_ws + off);
        off += (nflt * sizeof(float) + 255) & ~(size_t)255;
        return p;
    };

    auto conv = [&](const float* in, int wi, const float* resid, float* out,
                    int Cin, int Cout, int Hin, int Win, int ks, int stride,
                    int pad, int act, float slope) {
        int Hout = (Hin + 2 * pad - ks) / stride + 1;
        int Wout = (Win + 2 * pad - ks) / stride + 1;
        int npx  = Hout * Wout;
        dim3 grid((unsigned)cdiv(cdiv(npx, 32), 4), (unsigned)cdiv(Cout, 32), 1);
        if (ks == 1)
            conv_wmma<1><<<grid, dim3(128), 0, stream>>>(
                in, Wp(wi), Wp(wi + 1), resid, out,
                Cin, Cout, Hin, Win, Hout, Wout, stride, pad, act, slope);
        else if (ks == 2)
            conv_wmma<2><<<grid, dim3(128), 0, stream>>>(
                in, Wp(wi), Wp(wi + 1), resid, out,
                Cin, Cout, Hin, Win, Hout, Wout, stride, pad, act, slope);
        else
            conv_wmma<3><<<grid, dim3(128), 0, stream>>>(
                in, Wp(wi), Wp(wi + 1), resid, out,
                Cin, Cout, Hin, Win, Hout, Wout, stride, pad, act, slope);
    };

    auto resblock = [&](const float* xin, int w1i, int w2i, float* tmp, float* out,
                        int C, int H, int Wd) {
        conv(xin, w1i, nullptr, tmp, C, C, H, Wd, 3, 1, 1, 1, 0.2f);
        conv(tmp, w2i, xin,     out, C, C, H, Wd, 3, 1, 1, 0, 0.f);
    };

    auto rsa = [&](const float* xin, const float* offf, int wom, int wd, int wc1,
                   float* om, float* val, float* fea, float* out,
                   int C, int H, int Wd) {
        conv(offf, wom, nullptr, om, 32, 216, H, Wd, 3, 1, 1, 0, 0.f);
        int tot = 8 * 9 * H * Wd;
        deform_sample_k<<<cdiv(tot, 256), 256, 0, stream>>>(xin, om, val, C, H, Wd, 8);
        conv(val, wd, nullptr, fea, C * 9, C, H, Wd, 1, 1, 0, 1, 0.2f);   // contraction
        conv(fea, wc1, xin,    out, C,     C, H, Wd, 3, 1, 1, 0, 0.f);
    };

    auto offblk = [&](const float* xin, const float* lastOff, int w1, int w2, int w3,
                      float* cat, float* tmp, float* out, int Cin, int H, int Wd) {
        int HW = H * Wd;
        conv(xin, w1, nullptr, cat, Cin, 32, H, Wd, 3, 1, 1, 1, 0.2f);
        upsample2x_k<<<cdiv(32 * HW, 256), 256, 0, stream>>>(
            lastOff, cat + (size_t)32 * HW, 32, H / 2, Wd / 2, 2.0f);
        conv(cat, w2, nullptr, tmp, 64, 32, H, Wd, 3, 1, 1, 1, 0.2f);
        conv(tmp, w3, nullptr, out, 32, 32, H, Wd, 3, 1, 1, 1, 0.2f);
    };

    auto uplevel = [&](const float* d, const float* skip, int wupi, int wdci,
                       float* cat, float* u, int Cd, int Cu, int Hlo, int Wlo) {
        int Ho = 2 * Hlo, Wo = 2 * Wlo, HWo = Ho * Wo;
        conv_t2_k<<<cdiv(Cu * HWo, 256), 256, 0, stream>>>(
            d, Wp(wupi), Wp(wupi + 1), cat, Cd, Cu, Hlo, Wlo);
        hipMemcpyAsync(cat + (size_t)Cu * HWo, skip, (size_t)Cu * HWo * sizeof(float),
                       hipMemcpyDeviceToDevice, stream);
        conv(cat, wdci, nullptr, u, 2 * Cu, Cu, Ho, Wo, 1, 1, 0, 0, 0.f);
    };

    // ---- FOV attention head ------------------------------------------------
    float* ai  = alloc((size_t)32 * HW1);
    float* af  = alloc((size_t)32 * HW1);
    float* ai2 = alloc((size_t)32 * HW1);
    float* af2 = alloc((size_t)32 * HW1);
    float* att = alloc((size_t)32 * HW1);
    conv(x,                1, nullptr, ai,  3,  32, H1, W1, 3, 1, 1, 1, 0.1f);  // fov_i1
    conv(x + (size_t)3*HW1,5, nullptr, af,  2,  32, H1, W1, 3, 1, 1, 1, 0.1f);  // fov_f1
    conv(ai,               3, nullptr, ai2, 32, 32, H1, W1, 1, 1, 0, 1, 0.1f);  // fov_i2
    conv(af,               7, nullptr, af2, 32, 32, H1, W1, 1, 1, 0, 2, 0.1f);  // fov_f2 (+sigmoid)
    att_k<<<cdiv(32 * HW1, 256), 256, 0, stream>>>(ai2, af2, att, 32 * HW1);

    // ---- encoder -----------------------------------------------------------
    float* t1 = alloc((size_t)32 * HW1);  float* c1 = alloc((size_t)32 * HW1);
    resblock(att, 9, 11, t1, c1, 32, H1, W1);
    float* p1 = alloc((size_t)64 * HW2);
    conv(c1, 13, nullptr, p1, 32, 64, H1, W1, 2, 2, 0, 1, 0.2f);

    float* t2 = alloc((size_t)64 * HW2);  float* c2 = alloc((size_t)64 * HW2);
    resblock(p1, 15, 17, t2, c2, 64, H2, W2);
    float* p2 = alloc((size_t)128 * HW3);
    conv(c2, 19, nullptr, p2, 64, 128, H2, W2, 2, 2, 0, 1, 0.2f);

    float* t3 = alloc((size_t)128 * HW3); float* c3 = alloc((size_t)128 * HW3);
    resblock(p2, 21, 23, t3, c3, 128, H3, W3);
    float* p3 = alloc((size_t)256 * HW4);
    conv(c3, 25, nullptr, p3, 128, 256, H3, W3, 2, 2, 0, 1, 0.2f);

    float* t4 = alloc((size_t)256 * HW4); float* c4 = alloc((size_t)256 * HW4);
    resblock(p3, 27, 29, t4, c4, 256, H4, W4);

    // ---- level 4 offsets + RSA --------------------------------------------
    float* o4t = alloc((size_t)32 * HW4); float* o4 = alloc((size_t)32 * HW4);
    conv(c4,  31, nullptr, o4t, 256, 32, H4, W4, 3, 1, 1, 1, 0.2f);
    conv(o4t, 33, nullptr, o4,  32,  32, H4, W4, 3, 1, 1, 1, 0.2f);
    float* om4  = alloc((size_t)216 * HW4);
    float* val4 = alloc((size_t)2304 * HW4);
    float* fea4 = alloc((size_t)256 * HW4);
    float* d4   = alloc((size_t)256 * HW4);
    rsa(c4, o4, 35, 37, 39, om4, val4, fea4, d4, 256, H4, W4);

    // ---- decoder level 3 ---------------------------------------------------
    float* cat3 = alloc((size_t)256 * HW3); float* u3 = alloc((size_t)128 * HW3);
    uplevel(d4, c3, 41, 43, cat3, u3, 256, 128, H4, W4);
    float* oc3  = alloc((size_t)64 * HW3);  float* ot3 = alloc((size_t)32 * HW3);
    float* o3   = alloc((size_t)32 * HW3);
    offblk(u3, o4, 45, 47, 49, oc3, ot3, o3, 128, H3, W3);
    float* om3  = alloc((size_t)216 * HW3);
    float* val3 = alloc((size_t)1152 * HW3);
    float* fea3 = alloc((size_t)128 * HW3);
    float* d3   = alloc((size_t)128 * HW3);
    rsa(u3, o3, 51, 53, 55, om3, val3, fea3, d3, 128, H3, W3);

    // ---- decoder level 2 ---------------------------------------------------
    float* cat2 = alloc((size_t)128 * HW2); float* u2 = alloc((size_t)64 * HW2);
    uplevel(d3, c2, 57, 59, cat2, u2, 128, 64, H3, W3);
    float* oc2  = alloc((size_t)64 * HW2);  float* ot2 = alloc((size_t)32 * HW2);
    float* o2   = alloc((size_t)32 * HW2);
    offblk(u2, o3, 61, 63, 65, oc2, ot2, o2, 64, H2, W2);
    float* om2  = alloc((size_t)216 * HW2);
    float* val2 = alloc((size_t)576 * HW2);
    float* fea2 = alloc((size_t)64 * HW2);
    float* d2   = alloc((size_t)64 * HW2);
    rsa(u2, o2, 67, 69, 71, om2, val2, fea2, d2, 64, H2, W2);

    // ---- decoder level 1 ---------------------------------------------------
    float* cat1 = alloc((size_t)64 * HW1); float* u1 = alloc((size_t)32 * HW1);
    uplevel(d2, c1, 73, 75, cat1, u1, 64, 32, H2, W2);
    float* oc1  = alloc((size_t)64 * HW1);  float* ot1 = alloc((size_t)32 * HW1);
    float* o1   = alloc((size_t)32 * HW1);
    offblk(u1, o2, 77, 79, 81, oc1, ot1, o1, 32, H1, W1);
    float* om1  = alloc((size_t)216 * HW1);
    float* val1 = alloc((size_t)288 * HW1);
    float* fea1 = alloc((size_t)32 * HW1);
    float* d1   = alloc((size_t)32 * HW1);
    rsa(u1, o1, 83, 85, 87, om1, val1, fea1, d1, 32, H1, W1);

    // ---- kernel prediction + final conv -----------------------------------
    float* core  = alloc((size_t)75 * HW1);
    float* preds = alloc((size_t)12 * HW1);
    conv(d1, 89, nullptr, core, 32, 75, H1, W1, 1, 1, 0, 0, 0.f);
    for (int r = 1; r <= 4; ++r)
        kernel_pred_k<<<cdiv(3 * HW1, 256), 256, 0, stream>>>(
            x, core, preds, H1, W1, r, (r - 1) * 3);
    conv(preds, 91, nullptr, (float*)d_out, 12, 3, H1, W1, 3, 1, 1, 0, 0.f);
}
